// GATRNN_71597104824371
// MI455X (gfx1250) — matrix operand order repaired and checked by reading, not compiled, not run on hardware
//
#include <hip/hip_runtime.h>
#include <hip/hip_bf16.h>

// ---------------------------------------------------------------------------
// Shapes (fixed by the reference)
// ---------------------------------------------------------------------------
#define NN      20000
#define EE      320000
#define EP      (EE + NN)     // edges + self loops
#define HH      4
#define CC      64
#define FNODE   64
#define FEDGE   16
#define FW      8
#define TT      24
#define LHID    64
#define GDIM    256           // H*C = 4*LH = 256 (gate width too)
#define NEG_SLOPE 0.2f
#define BN_EPS    1e-5f

typedef __attribute__((ext_vector_type(16))) _Float16 v16h;
typedef __attribute__((ext_vector_type(8)))  float    v8f;

// ---------------------------------------------------------------------------
// Workspace layout (float offsets). Zero region = [0, ZERO_COUNT).
// ---------------------------------------------------------------------------
#define OFF_ACC     0u            // N*256 GAT accumulators
#define OFF_DEG     5120000u      // N degrees
#define OFF_MEAN    5140000u      // N*16 mean edge attr (accum then in-place mean)
#define OFF_AMAX    5460000u      // N*4 encoded max (uint)
#define OFF_DENOM   5540000u      // N*4 softmax denominators
#define OFF_BN      5620000u      // 64 sum + 64 sumsq
#define ZERO_COUNT  5620128u
#define OFF_XH      5620160u      // N*256 f32
#define OFF_ALPHA   10740160u     // (E+N)*4  alpha -> ex (reused)
#define OFF_ASRC    12100160u     // N*4
#define OFF_ADST    12180160u     // N*4
#define OFF_X1RAW   12260160u     // N*64
#define OFF_X1      13540160u     // N*64
#define OFF_HLAST   14820160u     // N*64
#define OFF_USRC    16100160u     // 64*4
#define OFF_UDST    16100416u     // 64*4
#define OFF_VEDGE   16100672u     // 16*4
#define OFF_PBLIN   16100736u     // 16384 halves (8192 f32 slots), 32B aligned
#define OFF_PBHH    16108928u     // 16384 halves

// ---------------------------------------------------------------------------
// Helpers
// ---------------------------------------------------------------------------
__device__ __forceinline__ float sigmoidf_(float x) { return 1.0f / (1.0f + expf(-x)); }

__device__ __forceinline__ unsigned fenc(float f) {
    unsigned b = __float_as_uint(f);
    return (b & 0x80000000u) ? ~b : (b | 0x80000000u);
}
__device__ __forceinline__ float fdec(unsigned k) {
    unsigned b = (k & 0x80000000u) ? (k ^ 0x80000000u) : ~k;
    return __uint_as_float(b);
}

// A fragment for v_wmma_f32_16x16x32_f16 from a row-major [16][64] f16 LDS tile.
// Lane L holds row m=L%16; per K-chunk kc: lanes 0-15 K={0..7,16..23}+32kc,
// lanes 16-31 K={8..15,24..31}+32kc  (ISA 7.12.2, 16-bit A 16x32 layout).
__device__ __forceinline__ v16h load_a_frag(const _Float16* a16, int lane, int kc) {
    int m    = lane & 15;
    int base = kc * 32 + ((lane & 16) ? 8 : 0);
    v16h r;
#pragma unroll
    for (int e = 0; e < 8; ++e) r[e]     = a16[m * 64 + base + e];
#pragma unroll
    for (int e = 0; e < 8; ++e) r[8 + e] = a16[m * 64 + base + 16 + e];
    return r;
}

// ---------------------------------------------------------------------------
// Zero the accumulator / stats region
// ---------------------------------------------------------------------------
__global__ void k_zero(float* ws) {
    unsigned i = blockIdx.x * blockDim.x + threadIdx.x;
    if (i < ZERO_COUNT) ws[i] = 0.0f;
}

// ---------------------------------------------------------------------------
// Pre-swizzle a B matrix (K=64 x N=256 f16) into WMMA fragment order:
// packed[((ct*2+kc)*32+lane)*16+e] = B[kc*32 + (lane>=16?8:0) + (e<8?e:e+8)][ct*16 + lane%16]
// (B layout: lane = column, K = 8h+e (e<8), 16+8h+(e-8) (e>=8) per lane half h; cf. ISA 7.12.4 B 64x16.)
// transpose=0: W is [64][256] (already K x N).  transpose=1: W is [256][64].
// ---------------------------------------------------------------------------
__global__ void k_pack_b(const float* __restrict__ W, _Float16* __restrict__ packed, int transpose) {
    int tid = blockIdx.x * blockDim.x + threadIdx.x;
    if (tid >= 16 * 2 * 32 * 16) return;
    int e    = tid & 15;
    int lane = (tid >> 4) & 31;
    int kc   = (tid >> 9) & 1;
    int ct   = tid >> 10;
    int k = kc * 32 + ((lane & 16) ? 8 : 0) + (e < 8 ? e : e + 8);
    int n = ct * 16 + (lane & 15);
    float v = transpose ? W[n * 64 + k] : W[k * GDIM + n];
    packed[tid] = (_Float16)v;
}

// ---------------------------------------------------------------------------
// Fold attention vectors through the projection weights:
//   u_src[f][h] = sum_c W_lin[f, h*64+c] * att_src[h][c]   (64x4), same u_dst
//   v_edge[f][h] = sum_c W_edge[f, h*64+c] * att_edge[h][c] (16x4)
// ---------------------------------------------------------------------------
__global__ void k_fold_att(const float* __restrict__ W_lin, const float* __restrict__ W_edge,
                           const float* __restrict__ att_src, const float* __restrict__ att_dst,
                           const float* __restrict__ att_edge,
                           float* __restrict__ u_src, float* __restrict__ u_dst,
                           float* __restrict__ v_edge) {
    int tid = threadIdx.x;
    if (tid < 256) {
        int f = tid >> 2, h = tid & 3;
        float ss = 0.f, sd = 0.f;
        for (int c = 0; c < CC; ++c) {
            float w = W_lin[f * GDIM + h * CC + c];
            ss += w * att_src[h * CC + c];
            sd += w * att_dst[h * CC + c];
        }
        u_src[tid] = ss;
        u_dst[tid] = sd;
        if (f < FEDGE) {
            float se = 0.f;
            for (int c = 0; c < CC; ++c)
                se += W_edge[f * GDIM + h * CC + c] * att_edge[h * CC + c];
            v_edge[tid] = se;
        }
    }
}

// ---------------------------------------------------------------------------
// Degree + edge-attr segment sum (atomics into zeroed buffers)
// ---------------------------------------------------------------------------
__global__ void k_deg_attr(const long long* __restrict__ ei, const float* __restrict__ eattr,
                           float* __restrict__ deg, float* __restrict__ mean) {
    int e = blockIdx.x * blockDim.x + threadIdx.x;
    if (e >= EE) return;
    int d = (int)ei[2 * e + 1];
    atomicAdd(&deg[d], 1.0f);
#pragma unroll
    for (int f = 0; f < FEDGE; ++f)
        atomicAdd(&mean[d * FEDGE + f], eattr[e * FEDGE + f]);
}

__global__ void k_mean(float* __restrict__ mean, const float* __restrict__ deg) {
    int i = blockIdx.x * blockDim.x + threadIdx.x;
    if (i >= NN * FEDGE) return;
    float dg = deg[i / FEDGE];
    mean[i] /= fmaxf(dg, 1.0f);
}

// ---------------------------------------------------------------------------
// xh = X @ W_lin  (N x 64 -> N x 256) via v_wmma_f32_16x16x32_f16.
// One wave per 16-node tile; 16 column tiles x 2 K-chunks.
// ---------------------------------------------------------------------------
__global__ __launch_bounds__(32) void k_xh_wmma(const float* __restrict__ X,
                                                const _Float16* __restrict__ packedB,
                                                float* __restrict__ xh) {
    __shared__ _Float16 a16[16 * 64];
    int lane = threadIdx.x;
    int row0 = blockIdx.x * 16;
    for (int i = lane; i < 16 * 64; i += 32)
        a16[i] = (_Float16)X[row0 * FNODE + i];
    __syncthreads();
    const v16h* pb = (const v16h*)packedB;
    int n  = lane & 15;
    int mb = (lane & 16) ? 8 : 0;
    for (int ct = 0; ct < 16; ++ct) {
        v8f acc = {};
#pragma unroll
        for (int kc = 0; kc < 2; ++kc) {
            v16h a = load_a_frag(a16, lane, kc);
            v16h b = pb[(ct * 2 + kc) * 32 + lane];
            acc = __builtin_amdgcn_wmma_f32_16x16x32_f16(false, a, false, b,
                                                         (short)0, acc, false, false);
        }
#pragma unroll
        for (int j = 0; j < 8; ++j)
            xh[(row0 + mb + j) * GDIM + ct * 16 + n] = acc[j];
    }
}

// ---------------------------------------------------------------------------
// Per-node attention scores a_src / a_dst
// ---------------------------------------------------------------------------
__global__ void k_node_scores(const float* __restrict__ X, const float* __restrict__ u_src,
                              const float* __restrict__ u_dst,
                              float* __restrict__ a_src, float* __restrict__ a_dst) {
    int nd = blockIdx.x * blockDim.x + threadIdx.x;
    if (nd >= NN) return;
    float ss[HH] = {0, 0, 0, 0}, sd[HH] = {0, 0, 0, 0};
    for (int f = 0; f < FNODE; ++f) {
        float x = X[nd * FNODE + f];
#pragma unroll
        for (int h = 0; h < HH; ++h) {
            ss[h] += x * u_src[f * HH + h];
            sd[h] += x * u_dst[f * HH + h];
        }
    }
#pragma unroll
    for (int h = 0; h < HH; ++h) {
        a_src[nd * HH + h] = ss[h];
        a_dst[nd * HH + h] = sd[h];
    }
}

// ---------------------------------------------------------------------------
// alpha = leaky_relu(a_src[src]+a_dst[dst]+a_edge); segment max via encoded u32
// ---------------------------------------------------------------------------
__global__ void k_edge_alpha(const long long* __restrict__ ei, const float* __restrict__ eattr,
                             const float* __restrict__ mean, const float* __restrict__ v_edge,
                             const float* __restrict__ a_src, const float* __restrict__ a_dst,
                             float* __restrict__ alpha, unsigned* __restrict__ amax) {
    int e = blockIdx.x * blockDim.x + threadIdx.x;
    if (e >= EP) return;
    int s, d;
    const float* attr;
    if (e < EE) { s = (int)ei[2 * e]; d = (int)ei[2 * e + 1]; attr = &eattr[e * FEDGE]; }
    else        { s = d = e - EE;                              attr = &mean[(size_t)d * FEDGE]; }
#pragma unroll
    for (int h = 0; h < HH; ++h) {
        float ae = 0.f;
        for (int f = 0; f < FEDGE; ++f) ae += attr[f] * v_edge[f * HH + h];
        float al = a_src[s * HH + h] + a_dst[d * HH + h] + ae;
        al = (al > 0.f) ? al : NEG_SLOPE * al;
        alpha[e * HH + h] = al;
        atomicMax(&amax[d * HH + h], fenc(al));
    }
}

// ex = exp(alpha - amax[dst]); denom = segment_sum(ex)
__global__ void k_edge_exp(const long long* __restrict__ ei, float* __restrict__ alpha,
                           const unsigned* __restrict__ amax, float* __restrict__ denom) {
    int e = blockIdx.x * blockDim.x + threadIdx.x;
    if (e >= EP) return;
    int d = (e < EE) ? (int)ei[2 * e + 1] : (e - EE);
#pragma unroll
    for (int h = 0; h < HH; ++h) {
        float ex = expf(alpha[e * HH + h] - fdec(amax[d * HH + h]));
        alpha[e * HH + h] = ex;
        atomicAdd(&denom[d * HH + h], ex);
    }
}

// msg = xh[src] * attn scattered into acc[dst]  (one block per edge, 256 lanes)
__global__ void k_scatter(const long long* __restrict__ ei, const float* __restrict__ xh,
                          const float* __restrict__ ex, const float* __restrict__ denom,
                          float* __restrict__ acc) {
    int e = blockIdx.x;
    int t = threadIdx.x;           // 0..255 : h = t/64, c = t%64
    int s, d;
    if (e < EE) { s = (int)ei[2 * e]; d = (int)ei[2 * e + 1]; }
    else        { s = d = e - EE; }
    int h = t >> 6;
    float w = ex[e * HH + h] / (denom[d * HH + h] + 1e-16f);
    atomicAdd(&acc[(size_t)d * GDIM + t], xh[(size_t)s * GDIM + t] * w);
}

// head-mean + bias, and BN statistics
__global__ void k_gat_stats(const float* __restrict__ acc, const float* __restrict__ gat_bias,
                            float* __restrict__ x1raw, float* __restrict__ bn) {
    int i = blockIdx.x * blockDim.x + threadIdx.x;
    if (i >= NN * CC) return;
    int nd = i / CC, c = i % CC;
    const float* a = &acc[(size_t)nd * GDIM];
    float v = (a[c] + a[CC + c] + a[2 * CC + c] + a[3 * CC + c]) * 0.25f + gat_bias[c];
    x1raw[i] = v;
    atomicAdd(&bn[c], v);
    atomicAdd(&bn[CC + c], v * v);
}

// batch-norm + relu
__global__ void k_bn_relu(const float* __restrict__ x1raw, const float* __restrict__ bn,
                          const float* __restrict__ gamma, const float* __restrict__ beta,
                          float* __restrict__ x1) {
    int i = blockIdx.x * blockDim.x + threadIdx.x;
    if (i >= NN * CC) return;
    int c = i % CC;
    float mu  = bn[c] * (1.0f / NN);
    float var = bn[CC + c] * (1.0f / NN) - mu * mu;
    float v = (x1raw[i] - mu) * rsqrtf(var + BN_EPS) * gamma[c] + beta[c];
    x1[i] = fmaxf(v, 0.0f);
}

// ---------------------------------------------------------------------------
// LSTM over T=24 steps. Recurrent GEMM h@W_hh^T (16 nodes x 64 -> 256) done
// with v_wmma_f32_16x16x32_f16 each step; input contribution (K=8) added
// scalar-wise. One wave per 16-node tile, all state in LDS.
// ---------------------------------------------------------------------------
__global__ __launch_bounds__(32) void k_lstm_wmma(const float* __restrict__ Xdyn,
                                                  const _Float16* __restrict__ packedBhh,
                                                  const float* __restrict__ W_ih,
                                                  const float* __restrict__ b_ih,
                                                  const float* __restrict__ b_hh,
                                                  float* __restrict__ h_last) {
    __shared__ _Float16 a16[16 * 64];
    __shared__ float g[16 * GDIM];
    __shared__ float hs[16 * LHID];
    __shared__ float cs[16 * LHID];
    __shared__ float wih[GDIM * FW];
    __shared__ float bias[GDIM];

    int lane = threadIdx.x;
    int row0 = blockIdx.x * 16;
    for (int i = lane; i < GDIM * FW; i += 32) wih[i] = W_ih[i];
    for (int i = lane; i < GDIM; i += 32)      bias[i] = b_ih[i] + b_hh[i];
    for (int i = lane; i < 16 * LHID; i += 32) { hs[i] = 0.f; cs[i] = 0.f; }
    __syncthreads();

    const v16h* pb = (const v16h*)packedBhh;
    int nc = lane & 15;
    int mb = (lane & 16) ? 8 : 0;

    for (int t = 0; t < TT; ++t) {
        for (int i = lane; i < 16 * LHID; i += 32) a16[i] = (_Float16)hs[i];
        __syncthreads();
        for (int ct = 0; ct < 16; ++ct) {
            v8f accv = {};
#pragma unroll
            for (int kc = 0; kc < 2; ++kc) {
                v16h a = load_a_frag(a16, lane, kc);
                v16h b = pb[(ct * 2 + kc) * 32 + lane];
                accv = __builtin_amdgcn_wmma_f32_16x16x32_f16(false, a, false, b,
                                                              (short)0, accv, false, false);
            }
#pragma unroll
            for (int j = 0; j < 8; ++j)
                g[(mb + j) * GDIM + ct * 16 + nc] = accv[j];
        }
        __syncthreads();
        for (int idx = lane; idx < 16 * LHID; idx += 32) {
            int m = idx >> 6, j = idx & 63;
            const float* x = &Xdyn[(size_t)(row0 + m) * TT * FW + (size_t)t * FW];
            float gi = g[m * GDIM + j]            + bias[j];
            float gf = g[m * GDIM + 64 + j]       + bias[64 + j];
            float gg = g[m * GDIM + 128 + j]      + bias[128 + j];
            float go = g[m * GDIM + 192 + j]      + bias[192 + j];
#pragma unroll
            for (int k = 0; k < FW; ++k) {
                float xv = x[k];
                gi += xv * wih[j * FW + k];
                gf += xv * wih[(64 + j) * FW + k];
                gg += xv * wih[(128 + j) * FW + k];
                go += xv * wih[(192 + j) * FW + k];
            }
            float c = sigmoidf_(gf) * cs[idx] + sigmoidf_(gi) * tanhf(gg);
            float h = sigmoidf_(go) * tanhf(c);
            cs[idx] = c;
            hs[idx] = h;
        }
        __syncthreads();
    }
    for (int i = lane; i < 16 * LHID; i += 32)
        h_last[(size_t)row0 * LHID + i] = hs[i];
}

// ---------------------------------------------------------------------------
// Final MLP: relu([x1, h_last] @ W_hid + b_hid) @ W_out + b_out
// ---------------------------------------------------------------------------
__global__ void k_mlp(const float* __restrict__ x1, const float* __restrict__ h_last,
                      const float* __restrict__ W_hid, const float* __restrict__ b_hid,
                      const float* __restrict__ W_out, const float* __restrict__ b_out,
                      float* __restrict__ out) {
    int nd = blockIdx.x * blockDim.x + threadIdx.x;
    if (nd >= NN) return;
    float o = 0.f;
    for (int j = 0; j < 32; ++j) {
        float s = b_hid[j];
        for (int k = 0; k < CC; ++k)   s += x1[nd * CC + k]     * W_hid[k * 32 + j];
        for (int k = 0; k < LHID; ++k) s += h_last[nd * LHID + k] * W_hid[(CC + k) * 32 + j];
        o += fmaxf(s, 0.f) * W_out[j];
    }
    out[nd] = o + b_out[0];
}

// ---------------------------------------------------------------------------
// Launch
// ---------------------------------------------------------------------------
extern "C" void kernel_launch(void* const* d_in, const int* in_sizes, int n_in,
                              void* d_out, int out_size, void* d_ws, size_t ws_size,
                              hipStream_t stream) {
    const float*     X      = (const float*)d_in[0];
    const float*     Eattr  = (const float*)d_in[1];
    const float*     Xdyn   = (const float*)d_in[2];
    const long long* EI     = (const long long*)d_in[3];
    const float*     W_lin  = (const float*)d_in[4];
    const float*     W_edge = (const float*)d_in[5];
    const float*     AttS   = (const float*)d_in[6];
    const float*     AttD   = (const float*)d_in[7];
    const float*     AttE   = (const float*)d_in[8];
    const float*     GBias  = (const float*)d_in[9];
    const float*     Gamma  = (const float*)d_in[10];
    const float*     Beta   = (const float*)d_in[11];
    const float*     W_ih   = (const float*)d_in[12];
    const float*     W_hh   = (const float*)d_in[13];
    const float*     B_ih   = (const float*)d_in[14];
    const float*     B_hh   = (const float*)d_in[15];
    const float*     W_hid  = (const float*)d_in[16];
    const float*     B_hid  = (const float*)d_in[17];
    const float*     W_out  = (const float*)d_in[18];
    const float*     B_out  = (const float*)d_in[19];
    float*           out    = (float*)d_out;

    float*     ws    = (float*)d_ws;
    float*     acc   = ws + OFF_ACC;
    float*     deg   = ws + OFF_DEG;
    float*     mean  = ws + OFF_MEAN;
    unsigned*  amax  = (unsigned*)(ws + OFF_AMAX);
    float*     denom = ws + OFF_DENOM;
    float*     bn    = ws + OFF_BN;
    float*     xh    = ws + OFF_XH;
    float*     alpha = ws + OFF_ALPHA;
    float*     a_src = ws + OFF_ASRC;
    float*     a_dst = ws + OFF_ADST;
    float*     x1raw = ws + OFF_X1RAW;
    float*     x1    = ws + OFF_X1;
    float*     hlast = ws + OFF_HLAST;
    float*     u_src = ws + OFF_USRC;
    float*     u_dst = ws + OFF_UDST;
    float*     vedge = ws + OFF_VEDGE;
    _Float16*  pblin = (_Float16*)(ws + OFF_PBLIN);
    _Float16*  pbhh  = (_Float16*)(ws + OFF_PBHH);

    k_zero<<<(ZERO_COUNT + 255) / 256, 256, 0, stream>>>(ws);
    k_pack_b<<<(16384 + 255) / 256, 256, 0, stream>>>(W_lin, pblin, 0);
    k_pack_b<<<(16384 + 255) / 256, 256, 0, stream>>>(W_hh,  pbhh,  1);
    k_fold_att<<<1, 256, 0, stream>>>(W_lin, W_edge, AttS, AttD, AttE, u_src, u_dst, vedge);
    k_deg_attr<<<(EE + 255) / 256, 256, 0, stream>>>(EI, Eattr, deg, mean);
    k_mean<<<(NN * FEDGE + 255) / 256, 256, 0, stream>>>(mean, deg);
    k_xh_wmma<<<NN / 16, 32, 0, stream>>>(X, pblin, xh);
    k_node_scores<<<(NN + 255) / 256, 256, 0, stream>>>(X, u_src, u_dst, a_src, a_dst);
    k_edge_alpha<<<(EP + 255) / 256, 256, 0, stream>>>(EI, Eattr, mean, vedge, a_src, a_dst, alpha, amax);
    k_edge_exp<<<(EP + 255) / 256, 256, 0, stream>>>(EI, alpha, amax, denom);
    k_scatter<<<EP, 256, 0, stream>>>(EI, xh, alpha, denom, acc);
    k_gat_stats<<<(NN * CC + 255) / 256, 256, 0, stream>>>(acc, GBias, x1raw, bn);
    k_bn_relu<<<(NN * CC + 255) / 256, 256, 0, stream>>>(x1raw, bn, Gamma, Beta, x1);
    k_lstm_wmma<<<NN / 16, 32, 0, stream>>>(Xdyn, pbhh, W_ih, B_ih, B_hh, hlast);
    k_mlp<<<(NN + 255) / 256, 256, 0, stream>>>(x1, hlast, W_hid, B_hid, W_out, B_out, out);

    (void)in_sizes; (void)n_in; (void)out_size; (void)ws_size;
}
